// GridMask_32727650795588
// MI455X (gfx1250) — compile-verified, matrix operand order checked by reading, not run on hardware
//
#include <hip/hip_runtime.h>

// GridMask for MI455X (gfx1250, wave32).
//
// Stage 1: build the 384x384 mask. The reference's covered[h,w] is an einsum
//          over 64 grid cells: rowInd[64,384]^T x colInd[64,384]. We compute it
//          with V_WMMA_F32_16X16X32_F16 (indicators are exact in f16; sums <=64
//          are exact in f32). One wave per 16x16 output tile, 2 K-steps (K=64).
// Stage 2: out = x * mask, broadcast over 384 planes. Pure streaming at
//          ~453 MB total -> ~19.5 us floor at 23.3 TB/s. float4 accesses,
//          non-temporal load/store for x/out so the cache-resident mask
//          (576 KB, 384x reuse) stays hot in WGP$/L2.

typedef __attribute__((ext_vector_type(16))) _Float16 v16h;
typedef __attribute__((ext_vector_type(8)))  float    v8f;
typedef __attribute__((ext_vector_type(4)))  float    f4;

#define H 384
#define W 384
#define MAXK 64          // gh*gw with d=48 -> 8*8
#define TILES_X (W / 16) // 24
#define TILES_Y (H / 16) // 24

// ---------------------------------------------------------------------------
// Stage 1: mask build via WMMA.
// Grid: (24*24 tiles) / (8 waves per 256-thread block) = 72 blocks.
// EXEC is all-ones in every wave (no divergence) as WMMA requires.
// ---------------------------------------------------------------------------
__global__ void __launch_bounds__(256)
gridmask_build_wmma(const int* __restrict__ cell_active,
                    const int* __restrict__ off_i,
                    const int* __restrict__ off_j,
                    const int* __restrict__ d_ptr,
                    float* __restrict__ mask) {
    __shared__ int s_ip[MAXK];
    __shared__ int s_jp[MAXK];
    __shared__ int s_act[MAXK];

    const int d  = d_ptr[0];
    const int l  = (int)((double)d * 0.6);   // matches Python int(d * 0.6)
    const int gh = (H + d - 1) / d;
    const int gw = (W + d - 1) / d;
    const int K  = gh * gw;                  // 64 for d=48; must be <= MAXK

    const int t = threadIdx.x;
    if (t < MAXK) {
        if (t < K) {
            const int a  = t / gw;
            const int bb = t % gw;
            int ip = a  * d + (off_i[t] - 2);
            int jp = bb * d + (off_j[t] - 2);
            ip = ip < 0 ? 0 : (ip > H - l ? H - l : ip);
            jp = jp < 0 ? 0 : (jp > W - l ? W - l : jp);
            s_ip[t]  = ip;
            s_jp[t]  = jp;
            s_act[t] = cell_active[t];
        } else {
            s_ip[t] = 0; s_jp[t] = 0; s_act[t] = 0;   // zero-pad K -> 64
        }
    }
    __syncthreads();

    const int wave = (blockIdx.x * blockDim.x + threadIdx.x) >> 5;
    const int lane = threadIdx.x & 31;
    const int ti = wave / TILES_X;
    const int tj = wave % TILES_X;

    // A-fragment lane mapping (16-bit A 16x32): m = lane%16,
    // K base 0 for lanes 0-15, 8 for lanes 16-31; halves e=0..15:
    //   v = e/2, p = e%2 ; klocal = v<4 ? kbase+2v+p : 16+kbase+2(v-4)+p
    // B-fragment mirrors A with n = lane%16.
    const int mn    = lane & 15;
    const int kbase = (lane < 16) ? 0 : 8;
    const int row   = ti * 16 + mn;   // A: matrix row index
    const int col   = tj * 16 + mn;   // B: matrix column index

    v8f acc = {};
#pragma unroll
    for (int kk = 0; kk < 2; ++kk) {               // two K=32 steps -> K=64
        v16h afrag, bfrag;
#pragma unroll
        for (int e = 0; e < 16; ++e) {
            const int v = e >> 1, p = e & 1;
            const int klocal = (v < 4) ? (kbase + 2 * v + p)
                                       : (16 + kbase + 2 * (v - 4) + p);
            const int k = kk * 32 + klocal;        // cell index 0..63
            const int ip = s_ip[k], jp = s_jp[k];
            // rfa: row indicator AND cell_active (matches reference)
            const bool aIn = (s_act[k] > 0) && (row >= ip) && (row < ip + l);
            // col_in: column indicator only
            const bool bIn = (col >= jp) && (col < jp + l);
            afrag[e] = aIn ? (_Float16)1.0f : (_Float16)0.0f;
            bfrag[e] = bIn ? (_Float16)1.0f : (_Float16)0.0f;
        }
        acc = __builtin_amdgcn_wmma_f32_16x16x32_f16(
            /*neg_a=*/false, afrag, /*neg_b=*/false, bfrag,
            /*c_mod=*/(short)0, acc, /*reuse_a=*/false, /*reuse_b=*/false);
    }

    // D layout (f32 16x16): lane n = lane%16, VGPR r -> M = r (+8 for lanes 16-31)
    const int n     = lane & 15;
    const int mbase = (lane < 16) ? 0 : 8;
#pragma unroll
    for (int r = 0; r < 8; ++r) {
        const int i = ti * 16 + mbase + r;
        const int j = tj * 16 + n;
        mask[i * W + j] = (acc[r] > 0.0f) ? 0.0f : 1.0f;
    }
}

// ---------------------------------------------------------------------------
// Stage 2: out = x * mask (broadcast over planes). Bandwidth bound.
// Each thread: one cached float4 mask load, then 8 planes of NT float4
// load/multiply/NT store (128B in + 128B out per thread).
// ---------------------------------------------------------------------------
__global__ void __launch_bounds__(256)
gridmask_apply(const float* __restrict__ x,
               const float* __restrict__ mask,
               const int* __restrict__ apply_flag,
               float* __restrict__ out,
               int planes) {
    const int PIX4 = (H * W) / 4;                          // 36864 float4 / plane
    const int q  = blockIdx.x * blockDim.x + threadIdx.x;  // [0, PIX4)
    const int p0 = blockIdx.y * 8;

    f4 m4 = ((const f4*)mask)[q];                          // cache-resident, RT
    if (apply_flag[0] <= 0) { m4.x = 1.f; m4.y = 1.f; m4.z = 1.f; m4.w = 1.f; }

#pragma unroll
    for (int pp = 0; pp < 8; ++pp) {
        const int p = p0 + pp;
        if (p < planes) {
            const long idx = (long)p * PIX4 + q;
            f4 xv = __builtin_nontemporal_load(((const f4*)x) + idx);
            f4 ov;
            ov.x = xv.x * m4.x;
            ov.y = xv.y * m4.y;
            ov.z = xv.z * m4.z;
            ov.w = xv.w * m4.w;
            __builtin_nontemporal_store(ov, ((f4*)out) + idx);
        }
    }
}

// ---------------------------------------------------------------------------
extern "C" void kernel_launch(void* const* d_in, const int* in_sizes, int n_in,
                              void* d_out, int out_size, void* d_ws, size_t ws_size,
                              hipStream_t stream) {
    const float* x           = (const float*)d_in[0];
    const int*   cell_active = (const int*)d_in[1];
    const int*   off_i       = (const int*)d_in[2];
    const int*   off_j       = (const int*)d_in[3];
    const int*   d_period    = (const int*)d_in[4];
    const int*   apply_flag  = (const int*)d_in[5];
    float*       out         = (float*)d_out;
    float*       mask        = (float*)d_ws;   // H*W floats = 576 KB scratch

    // Stage 1: 576 tiles, 1 wave each, 8 waves/block -> 72 blocks of 256.
    gridmask_build_wmma<<<dim3((TILES_Y * TILES_X) / 8), 256, 0, stream>>>(
        cell_active, off_i, off_j, d_period, mask);

    // Stage 2: grid over (within-plane float4 groups, plane groups of 8).
    const int planes = in_sizes[0] / (H * W);              // 128*3 = 384
    dim3 grid((H * W / 4) / 256, (planes + 7) / 8);        // (144, 48)
    gridmask_apply<<<grid, 256, 0, stream>>>(x, mask, apply_flag, out, planes);
}
